// MultiHeadAttention_10574209482796
// MI455X (gfx1250) — matrix operand order repaired and checked by reading, not compiled, not run on hardware
//
#include <hip/hip_runtime.h>

// MI455X / gfx1250, wave32. All matmuls via v_wmma_f32_16x16x32_bf16.
// Pipeline: 3x (X @ W^T + b) -> bf16 heads layout; flash-attention; ctx @ Wo^T + bo -> f32.
// Data movement: async global->LDS (ASYNCcnt) for bf16 tiles, packed cvt for f32->bf16 staging.

typedef __attribute__((ext_vector_type(16))) __bf16 v16bf;
typedef __attribute__((ext_vector_type(8)))  float  v8f;
typedef int v4i_ __attribute__((vector_size(16)));   // matches builtin param type V4i

union Frag { v16bf v; unsigned u[8]; };
union Acc  { v8f  v; float    f[8]; };

constexpr int Bb = 4, Ss = 2048, Dm = 1024, Hh = 16, DKk = 64;
constexpr int Mtok = Bb * Ss;
constexpr int LSTR = 72;   // LDS row stride in shorts: 144B = 9*16B (16B-aligned chunks, banks spread)

__device__ __forceinline__ unsigned short f2bf(float f) {
    union { float f; unsigned u; } x; x.f = f;
    unsigned r = x.u + 0x7FFFu + ((x.u >> 16) & 1u);   // round-nearest-even
    return (unsigned short)(r >> 16);
}

__device__ __forceinline__ unsigned f2bf2(float lo, float hi) {
#if __has_builtin(__builtin_amdgcn_cvt_pk_bf16_f32)
    typedef __attribute__((ext_vector_type(2))) __bf16 v2bf;
    union { v2bf v; unsigned u; } r;
    r.v = __builtin_amdgcn_cvt_pk_bf16_f32(lo, hi);
    return r.u;
#else
    return (unsigned)f2bf(lo) | ((unsigned)f2bf(hi) << 16);
#endif
}

// Async copy of one 16-byte chunk global->LDS (CDNA5 GLOBAL_LOAD_ASYNC_TO_LDS_B128, ASYNCcnt).
__device__ __forceinline__ void async_b128(const unsigned short* g, unsigned short* l) {
#if __has_builtin(__builtin_amdgcn_global_load_async_to_lds_b128)
    __builtin_amdgcn_global_load_async_to_lds_b128(
        (__attribute__((address_space(1))) v4i_*)(unsigned long long)(size_t)g,
        (__attribute__((address_space(3))) v4i_*)(unsigned)(size_t)l, 0, 0);
#else
    unsigned long long ga = (unsigned long long)(size_t)g;
    unsigned la = (unsigned)(size_t)l;
    asm volatile("global_load_async_to_lds_b128 %0, %1, off" :: "v"(la), "v"(ga) : "memory");
#endif
}

__device__ __forceinline__ void wait_async0() {
#if __has_builtin(__builtin_amdgcn_s_wait_asynccnt)
    __builtin_amdgcn_s_wait_asynccnt(0);
#else
    asm volatile("s_wait_asynccnt 0" ::: "memory");
#endif
}

// ---------------------------------------------------------------------------
// GEMM: C[m,n] = sum_k A[m,k] * W[n,k] + bias[n]      (torch Linear: x @ W^T + b)
// 256 threads (8 waves), 128x128 tile, staged K-step 64 (2 WMMA k-slices per stage).
// Wave (wm 0..1, wn 0..3) owns 64 rows x 32 cols = 4x2 WMMA tiles -> 16 wmma/stage.
// ---------------------------------------------------------------------------
template <bool A_BF16, bool OUT_HEADS>
__global__ __launch_bounds__(256)
void gemm_bias_kernel(const void* __restrict__ Ap, const float* __restrict__ Wp,
                      const float* __restrict__ bias, void* __restrict__ outp)
{
    constexpr int K = Dm, N = Dm;
    __shared__ unsigned short lds_a[128 * LSTR];
    __shared__ unsigned short lds_b[128 * LSTR];

    const int tid  = threadIdx.x;
    const int wave = tid >> 5, lane = tid & 31;
    const int hf   = lane >> 4, lm = lane & 15;
    const int wm   = wave >> 2, wn = wave & 3;
    const int m0   = blockIdx.x * 128, n0 = blockIdx.y * 128;

    const float*          Af = (const float*)Ap;
    const unsigned short* Ab = (const unsigned short*)Ap;

    Acc acc[4][2];
    for (int i = 0; i < 4; i++)
        for (int j = 0; j < 2; j++)
            for (int e = 0; e < 8; e++) acc[i][j].f[e] = 0.0f;

    for (int k0 = 0; k0 < K; k0 += 64) {
        // ---- stage A tile 128x64 ----
        if (A_BF16) {
            // already bf16: async DMA 16B chunks straight into LDS
            for (int i = 0; i < 4; i++) {
                int ch = tid + 256 * i;                 // 1024 chunks
                int r = ch >> 3, cc = ch & 7;
                async_b128(Ab + (size_t)(m0 + r) * K + k0 + cc * 8,
                           &lds_a[r * LSTR + cc * 8]);
            }
        } else {
            for (int i = 0; i < 16; i++) {
                int e = tid + 256 * i;                  // 4096 float pairs
                int r = e >> 5, c2 = e & 31;
                const float2 v2 = ((const float2*)(Af + (size_t)(m0 + r) * K + k0))[c2];
                ((unsigned*)&lds_a[r * LSTR])[c2] = f2bf2(v2.x, v2.y);
            }
        }
        // ---- stage B tile: lds_b[n][k] = W[n0+n][k0+k] (fp32 -> bf16) ----
        for (int i = 0; i < 16; i++) {
            int e = tid + 256 * i;
            int r = e >> 5, c2 = e & 31;
            const float2 v2 = ((const float2*)(Wp + (size_t)(n0 + r) * K + k0))[c2];
            ((unsigned*)&lds_b[r * LSTR])[c2] = f2bf2(v2.x, v2.y);
        }
        if (A_BF16) wait_async0();
        __syncthreads();

        for (int ks = 0; ks < 2; ks++) {
            Frag a[4], b[2];
            for (int sm = 0; sm < 4; sm++) {
                const unsigned* p = (const unsigned*)&lds_a[(wm * 64 + sm * 16 + lm) * LSTR];
                for (int j = 0; j < 4; j++) {
                    a[sm].u[j]     = p[ks * 16 + hf * 4 + j];
                    a[sm].u[4 + j] = p[ks * 16 + 8 + hf * 4 + j];
                }
            }
            for (int sn = 0; sn < 2; sn++) {
                const unsigned* p = (const unsigned*)&lds_b[(wn * 32 + sn * 16 + lm) * LSTR];
                for (int j = 0; j < 8; j++) b[sn].u[j] = p[ks * 16 + (j >> 2) * 8 + hf * 4 + (j & 3)];
            }
            for (int sm = 0; sm < 4; sm++)
                for (int sn = 0; sn < 2; sn++)
                    acc[sm][sn].v = __builtin_amdgcn_wmma_f32_16x16x32_bf16(
                        false, a[sm].v, false, b[sn].v, (short)0, acc[sm][sn].v, false, false);
        }
        __syncthreads();
    }

    // store: C layout -> lane holds col n=lm, rows v + 8*hf
    for (int sn = 0; sn < 2; sn++) {
        int n_g = n0 + wn * 32 + sn * 16 + lm;
        float bv = bias[n_g];
        for (int sm = 0; sm < 4; sm++) {
            for (int v = 0; v < 8; v++) {
                int m_g   = m0 + wm * 64 + sm * 16 + v + 8 * hf;
                float val = acc[sm][sn].f[v] + bv;
                if (OUT_HEADS) {
                    int b = m_g / Ss, s = m_g % Ss;
                    int h = n_g >> 6, dk = n_g & 63;
                    ((unsigned short*)outp)[(((size_t)(b * Hh + h)) * Ss + s) * DKk + dk] = f2bf(val);
                } else {
                    ((float*)outp)[(size_t)m_g * N + n_g] = val;
                }
            }
        }
    }
}

// ---------------------------------------------------------------------------
// Flash attention: block = one (b,h) x 64 query rows; 4 waves, each 16 rows.
// Q/K/V bf16 [B,H,S,64] -> ctx bf16 [B,S,1024].
// ---------------------------------------------------------------------------
__global__ __launch_bounds__(128)
void attn_kernel(const unsigned short* __restrict__ Qb,
                 const unsigned short* __restrict__ Kb,
                 const unsigned short* __restrict__ Vb,
                 unsigned short* __restrict__ Ctx)
{
    __shared__ unsigned short lds_k[64 * LSTR];   // [key][dk]  (B layout for Q*K^T)
    __shared__ unsigned short lds_v[64 * LSTR];   // [dk][key]  (B layout for P*V)
    __shared__ unsigned short lds_p[64 * LSTR];   // 4 waves x 16 rows x 64 keys

    const int tid  = threadIdx.x;
    const int wave = tid >> 5, lane = tid & 31;
    const int hf   = lane >> 4, lm = lane & 15;
    const int bh   = blockIdx.x;                  // b*H + h
    const int q0   = blockIdx.y * 64 + wave * 16;

    const size_t head_base = (size_t)bh * Ss * DKk;

    // Hoisted Q A-fragments for dk-steps 0,1 (rows contiguous: 64 bf16)
    Frag qa[2];
    {
        const unsigned* qrow = (const unsigned*)(Qb + head_base + (size_t)(q0 + lm) * DKk);
        for (int ks = 0; ks < 2; ks++)
            for (int j = 0; j < 4; j++) {
                qa[ks].u[j]     = qrow[ks * 16 + hf * 4 + j];
                qa[ks].u[4 + j] = qrow[ks * 16 + 8 + hf * 4 + j];
            }
    }

    Acc o[4];
    for (int t = 0; t < 4; t++)
        for (int e = 0; e < 8; e++) o[t].f[e] = 0.0f;
    float mrow[8], lrow[8];
    for (int v = 0; v < 8; v++) { mrow[v] = -3.0e38f; lrow[v] = 0.0f; }

    for (int j0 = 0; j0 < Ss; j0 += 64) {
        // stage K tile [key][dk] via async DMA (8KB contiguous in global)
        for (int i = 0; i < 4; i++) {
            int ch = tid + 128 * i;                  // 512 chunks of 16B
            int key = ch >> 3, cc = ch & 7;
            async_b128(Kb + head_base + (size_t)(j0 + key) * DKk + cc * 8,
                       &lds_k[key * LSTR + cc * 8]);
        }
        // stage V tile transposed [dk][key] (uint load, two b16 stores)
        {
            const unsigned* srcu = (const unsigned*)(Vb + head_base + (size_t)j0 * DKk);
            for (int i = 0; i < 16; i++) {
                int e = tid + 128 * i;               // 2048 uints
                int key = e >> 5, c2 = e & 31;
                unsigned u = srcu[key * 32 + c2];
                lds_v[(2 * c2) * LSTR + key]     = (unsigned short)u;
                lds_v[(2 * c2 + 1) * LSTR + key] = (unsigned short)(u >> 16);
            }
        }
        // prefetch next K/V tile rows (global_prefetch_b8)
        if (j0 + 64 < Ss) {
            int pk = tid & 63;
            __builtin_prefetch(Kb + head_base + (size_t)(j0 + 64 + pk) * DKk, 0, 1);
            __builtin_prefetch(Vb + head_base + (size_t)(j0 + 64 + pk) * DKk, 0, 1);
        }
        wait_async0();
        __syncthreads();

        // scores: S = Q (16x64) * K^T, 4 key-col tiles x 2 dk-steps
        Acc s[4];
        for (int nt = 0; nt < 4; nt++) {
            for (int e = 0; e < 8; e++) s[nt].f[e] = 0.0f;
            for (int ks = 0; ks < 2; ks++) {
                Frag kb;
                const unsigned* p = (const unsigned*)&lds_k[(nt * 16 + lm) * LSTR];
                for (int j = 0; j < 8; j++) kb.u[j] = p[ks * 16 + (j >> 2) * 8 + hf * 4 + (j & 3)];
                s[nt].v = __builtin_amdgcn_wmma_f32_16x16x32_bf16(
                    false, qa[ks].v, false, kb.v, (short)0, s[nt].v, false, false);
            }
        }
        for (int nt = 0; nt < 4; nt++)
            for (int v = 0; v < 8; v++) s[nt].f[v] *= 0.125f;   // 1/sqrt(64)

        // online softmax (row = v + 8*hf; 16 key-cols striped over lanes of the half)
        float mnew[8];
        for (int v = 0; v < 8; v++) {
            float x = fmaxf(fmaxf(s[0].f[v], s[1].f[v]), fmaxf(s[2].f[v], s[3].f[v]));
            for (int msk = 1; msk < 16; msk <<= 1) x = fmaxf(x, __shfl_xor(x, msk, 32));
            mnew[v] = fmaxf(mrow[v], x);
        }
        for (int v = 0; v < 8; v++) {
            float alpha = __expf(mrow[v] - mnew[v]);
            mrow[v] = mnew[v];
            float rs = 0.0f;
            for (int nt = 0; nt < 4; nt++) {
                float p = __expf(s[nt].f[v] - mnew[v]);
                rs += p;
                lds_p[(wave * 16 + v + 8 * hf) * LSTR + nt * 16 + lm] = f2bf(p);
            }
            for (int msk = 1; msk < 16; msk <<= 1) rs += __shfl_xor(rs, msk, 32);
            lrow[v] = lrow[v] * alpha + rs;
            for (int t = 0; t < 4; t++) o[t].f[v] *= alpha;
        }
        // same-wave LDS write->read ordering for P round-trip
        asm volatile("s_wait_dscnt 0" ::: "memory");

        // O += P (16 x 64keys) * V (64keys x 64dk)
        for (int kk = 0; kk < 2; kk++) {
            Frag pa;
            const unsigned* pp = (const unsigned*)&lds_p[(wave * 16 + lm) * LSTR];
            for (int j = 0; j < 4; j++) {
                pa.u[j]     = pp[kk * 16 + hf * 4 + j];
                pa.u[4 + j] = pp[kk * 16 + 8 + hf * 4 + j];
            }
            for (int nt = 0; nt < 4; nt++) {
                Frag vb;
                const unsigned* vp = (const unsigned*)&lds_v[(nt * 16 + lm) * LSTR];
                for (int j = 0; j < 8; j++) vb.u[j] = vp[kk * 16 + (j >> 2) * 8 + hf * 4 + (j & 3)];
                o[nt].v = __builtin_amdgcn_wmma_f32_16x16x32_bf16(
                    false, pa.v, false, vb.v, (short)0, o[nt].v, false, false);
            }
        }
        __syncthreads();
    }

    // normalize and scatter into ctx [B,S,1024]
    const int b = bh / Hh, h = bh % Hh;
    for (int nt = 0; nt < 4; nt++) {
        for (int v = 0; v < 8; v++) {
            int srow  = q0 + v + 8 * hf;
            float val = o[nt].f[v] / lrow[v];
            Ctx[((size_t)(b * Ss + srow)) * Dm + h * DKk + nt * 16 + lm] = f2bf(val);
        }
    }
}

// ---------------------------------------------------------------------------
extern "C" void kernel_launch(void* const* d_in, const int* in_sizes, int n_in,
                              void* d_out, int out_size, void* d_ws, size_t ws_size,
                              hipStream_t stream)
{
    (void)in_sizes; (void)n_in; (void)out_size; (void)ws_size;

    const float* q  = (const float*)d_in[0];
    const float* k  = (const float*)d_in[1];
    const float* v  = (const float*)d_in[2];
    const float* Wq = (const float*)d_in[3];
    const float* bq = (const float*)d_in[4];
    const float* Wk = (const float*)d_in[5];
    const float* bk = (const float*)d_in[6];
    const float* Wv = (const float*)d_in[7];
    const float* bv = (const float*)d_in[8];
    const float* Wo = (const float*)d_in[9];
    const float* bo = (const float*)d_in[10];

    unsigned short* ws = (unsigned short*)d_ws;
    const size_t TD = (size_t)Mtok * Dm;
    unsigned short* qb = ws;                      // Q bf16  [B,H,S,64]
    unsigned short* kb = ws + TD;                 // K bf16  [B,H,S,64]
    unsigned short* vb = ws + 2 * TD;             // V bf16  [B,H,S,64]
    unsigned short* cb = ws + 3 * TD;             // ctx bf16 [B,S,1024]

    dim3 gg(Mtok / 128, Dm / 128);                // 64 x 8 blocks
    gemm_bias_kernel<false, true><<<gg, 256, 0, stream>>>(q, Wq, bq, qb);
    gemm_bias_kernel<false, true><<<gg, 256, 0, stream>>>(k, Wk, bk, kb);
    gemm_bias_kernel<false, true><<<gg, 256, 0, stream>>>(v, Wv, bv, vb);

    dim3 ga(Bb * Hh, Ss / 64);                    // 64 x 32 blocks
    attn_kernel<<<ga, 128, 0, stream>>>(qb, kb, vb, cb);

    gemm_bias_kernel<true, false><<<gg, 256, 0, stream>>>(cb, Wo, bo, (float*)d_out);
}
